// ShiftNMF_59330678227046
// MI455X (gfx1250) — matrix operand order repaired and checked by reading, not compiled, not run on hardware
//
#include <hip/hip_runtime.h>
#include <math.h>

typedef __attribute__((ext_vector_type(16))) _Float16 v16h;
typedef __attribute__((ext_vector_type(8)))  float    v8f;

#define DIM   8      // reduction dim d
#define SPLIT 8      // K-loop split for DFT occupancy

__device__ __forceinline__ float softplusf(float x) {
    // jax.nn.softplus = log1p(exp(x)); stable for large x
    return x > 20.0f ? x : log1pf(expf(x));
}

// ---------------------------------------------------------------------------
// Kernel 0: pack softplus(H) directly into the per-lane WMMA A-fragment
// layout.  Apk[((t*32 + lane)*16) + e] = A element for K-tile t, lane, elem e.
// A 16x32 layout used here (contiguous k, same order in B): row = lane&15,  K = 16*(lane>>4) + e.
// Rows 8..15 and m >= M are zero (padding), so the DFT loop needs no bounds
// checks at all.
// ---------------------------------------------------------------------------
__global__ void k_pack(const float* __restrict__ H, _Float16* __restrict__ Apk,
                       int M, int mtiles) {
    int i = blockIdx.x * blockDim.x + threadIdx.x;
    if (i >= mtiles * 512) return;          // 32 lanes * 16 elems per tile
    int e    = i & 15;
    int lane = (i >> 4) & 31;
    int t    = i >> 9;
    int row  = lane & 15;
    int mA   = t * 32 + ((lane >> 4) << 4) + e;
    float v  = 0.0f;
    if (row < DIM && mA < M) v = softplusf(H[row * M + mA]);
    Apk[i] = (_Float16)v;
}

// ---------------------------------------------------------------------------
// Kernel 1: direct DFT via WMMA, D(16x16) += A(16x32,f16) x B(32x16,f16).
//   Output columns interleave Re/Im of 8 complex k's per tile:
//     B[m,2j]   =  cos(2*pi*k_j*m/M)
//     B[m,2j+1] = -sin(2*pi*k_j*m/M)         (e^{-i*theta})
//   Phase index r = (k*m) mod M is maintained incrementally -> no divisions.
//   cos folded into sin via +pi/2 offset -> uniform control flow, 1 trans/elem.
// ---------------------------------------------------------------------------
__global__ void __launch_bounds__(32)
k_dft_wmma(const _Float16* __restrict__ Apk, float* __restrict__ Hpart,
           int M, int mtiles) {
    const int lane  = threadIdx.x;          // wave32
    const int split = blockIdx.x;           // 0..SPLIT-1
    const int ktile = blockIdx.y;           // 0..ceil(M/8)-1

    const int  ncol  = lane & 15;           // output column / B column
    const int  khalf = lane >> 4;
    const int  k     = ktile * 8 + (ncol >> 1);
    const int  kk    = (k < M) ? k : 0;     // clamp for math on last tile
    const bool isIm  = (ncol & 1) != 0;

    const float w0  = 6.28318530717958647692f / (float)M;
    const float off = isIm ? 0.0f : 1.57079632679489662f;   // cos(x)=sin(x+pi/2)
    const float sgn = isIm ? -1.0f : 1.0f;

    const int chunk = (mtiles + SPLIT - 1) / SPLIT;
    const int t0 = split * chunk;
    const int t1 = (t0 + chunk < mtiles) ? (t0 + chunk) : mtiles;

    // r0 = (kk * m0) mod M for this lane's first B element; one mod total.
    int r0     = (int)(((long long)kk * (long long)(t0 * 32 + khalf * 16))
                       % (long long)M);
    int step32 = (int)(((long long)kk * 32ll) % (long long)M);

    v8f acc = {};
    for (int t = t0; t < t1; ++t) {
        // whole A fragment in one aligned 32-byte vector load
        v16h a = *(const v16h*)(Apk + (((size_t)t * 32 + lane) << 4));
        v16h b;
        int r = r0;
#pragma unroll
        for (int e = 0; e < 16; ++e) {
            float th = fmaf(w0, (float)r, off);
            b[e] = (_Float16)(sgn * __sinf(th));
            r += kk;
            if (r >= M) r -= M;
        }
        r0 += step32;
        if (r0 >= M) r0 -= M;
        acc = __builtin_amdgcn_wmma_f32_16x16x32_f16(
            /*neg_a=*/false, a, /*neg_b=*/false, b,
            /*c_mod=*/(short)0, acc, /*reuse_a=*/false, /*reuse_b=*/false);
    }

    // D layout: lanes 0-15 carry N=lane, VGPR v carries M=v (= source row d)
    if (lane < 16 && k < M) {
        float* dst = Hpart + (size_t)split * (size_t)(DIM * M * 2);
        const int comp = lane & 1;
#pragma unroll
        for (int v = 0; v < DIM; ++v)
            dst[((size_t)v * M + k) * 2 + comp] = acc[v];
    }
}

// ---------------------------------------------------------------------------
// Kernel 1b: deterministic reduction of the SPLIT partial DFTs.
// ---------------------------------------------------------------------------
__global__ void k_reduce(const float* __restrict__ Hpart, float* __restrict__ Hft,
                         int count) {
    int i = blockIdx.x * blockDim.x + threadIdx.x;
    if (i < count) {
        float s = 0.0f;
#pragma unroll
        for (int p = 0; p < SPLIT; ++p) s += Hpart[(size_t)p * count + i];
        Hft[i] = s;
    }
}

// ---------------------------------------------------------------------------
// Kernel 2: V[n,m] = sum_d W[n,d] * e^{-i*2pi*tau[n,d]*m/M} * Hft[d,m]
// Memory-bound: 40 MB complex output -> coalesced float2 loads/stores.
// ---------------------------------------------------------------------------
__global__ void k_einsum(const float* __restrict__ W, const float* __restrict__ tau,
                         const float2* __restrict__ Hft, float2* __restrict__ out,
                         int N, int M) {
    const int m = blockIdx.x * blockDim.x + threadIdx.x;
    const int n = blockIdx.y;
    if (m >= M) return;

    const float fm = (float)m / (float)M;
    float vre = 0.0f, vim = 0.0f;
#pragma unroll
    for (int d = 0; d < DIM; ++d) {
        float w  = W[n * DIM + d];
        float ph = -6.28318530717958647692f * tau[n * DIM + d] * fm;
        float s, c;
        __sincosf(ph, &s, &c);              // e^{-i*2pi*tau*f} = c + i*s
        float2 h = Hft[(size_t)d * M + m];
        vre += w * (c * h.x - s * h.y);
        vim += w * (c * h.y + s * h.x);
    }
    float2 res;
    res.x = vre;
    res.y = vim;
    out[(size_t)n * M + m] = res;
}

// ---------------------------------------------------------------------------
extern "C" void kernel_launch(void* const* d_in, const int* in_sizes, int n_in,
                              void* d_out, int out_size, void* d_ws, size_t ws_size,
                              hipStream_t stream) {
    const float* W   = (const float*)d_in[0];   // (N, 8)
    const float* H   = (const float*)d_in[1];   // (8, M)
    const float* tau = (const float*)d_in[2];   // (N, 8)

    const int N = in_sizes[0] / DIM;
    const int M = in_sizes[1] / DIM;

    const int mtiles = (M + 31) / 32;    // K-dim tiles of 32
    const int ktiles = (M + 7) / 8;      // 8 complex outputs per WMMA tile

    // workspace: [ Apk f16 (mtiles*512) | Hpart f32 (SPLIT*8*M*2) | Hft f32 (8*M*2) ]
    char* ws = (char*)d_ws;
    size_t apkBytes  = ((size_t)mtiles * 512 * sizeof(_Float16) + 255) & ~(size_t)255;
    size_t partBytes = ((size_t)SPLIT * DIM * M * 2 * sizeof(float) + 255) & ~(size_t)255;
    _Float16* Apk   = (_Float16*)ws;
    float*    Hpart = (float*)(ws + apkBytes);
    float*    Hft   = (float*)(ws + apkBytes + partBytes);

    k_pack<<<(mtiles * 512 + 255) / 256, 256, 0, stream>>>(H, Apk, M, mtiles);
    k_dft_wmma<<<dim3(SPLIT, ktiles), 32, 0, stream>>>(Apk, Hpart, M, mtiles);
    k_reduce<<<(DIM * M * 2 + 255) / 256, 256, 0, stream>>>(Hpart, Hft, DIM * M * 2);
    k_einsum<<<dim3((M + 255) / 256, N), 256, 0, stream>>>(
        W, tau, (const float2*)Hft, (float2*)d_out, N, M);
}
